// MyRNN_62113817035024
// MI455X (gfx1250) — matrix operand order repaired and checked by reading, not compiled, not run on hardware
//
#include <hip/hip_runtime.h>
#include <hip/hip_bf16.h>

// MI455X / gfx1250: wave32, WMMA bf16 16x16x32 with f32 accumulate.
//
// Strategy: 2-layer LSTM, B=256 T=256 E=256 U=512. Weights (7 MB bf16) are
// L2-resident (192 MB L2) so the 256 sequential steps are latency-bound, not
// HBM-bound. All conversions (embedding gather f32->bf16, weight transpose)
// are hoisted into fully parallel prep kernels; the per-step kernels contain
// only bf16 loads + v_wmma + gate TRANS math.

typedef __attribute__((ext_vector_type(16))) __bf16 v16bf;
typedef __attribute__((ext_vector_type(8)))  __bf16 v8bf;
typedef __attribute__((ext_vector_type(8)))  float  v8f;

#define B_  256
#define T_  256
#define E_  256
#define U_  512
#define G4  2048  // 4*U

__device__ __forceinline__ __bf16 f2bf(float f) {
  unsigned u = __builtin_bit_cast(unsigned, f);
  u += 0x7FFFu + ((u >> 16) & 1u);                 // round-to-nearest-even
  unsigned short s = (unsigned short)(u >> 16);
  return __builtin_bit_cast(__bf16, s);
}

__device__ __forceinline__ float sigf(float x) { return 1.0f / (1.0f + expf(-x)); }

__device__ __forceinline__ v8f wmma_bf16(v16bf a, v16bf b, v8f c) {
  // (neg_a, A, neg_b, B, c_mod, C, reuse_a, reuse_b)
  return __builtin_amdgcn_wmma_f32_16x16x32_bf16(false, a, false, b, (short)0, c,
                                                 false, false);
}

// A-fragment (16x32 bf16): lane L -> row L&15; K = (L>>4)*16 + {0..15}
__device__ __forceinline__ v16bf load_a_bf(const __bf16* p) {
  v8bf lo = *(const v8bf*)(p);
  v8bf hi = *(const v8bf*)(p + 8);
  return __builtin_shufflevector(lo, hi, 0,1,2,3,4,5,6,7,8,9,10,11,12,13,14,15);
}

// --------------------------------------------------------------------------
// Prep 1: weights f32 [K][N] -> bf16 [N][K] so the B-fragment is one
// contiguous 32B load per lane (B layout: lane L -> col L&15, K=(L>>4)*16+0..15).
__global__ void conv_transpose_bf16(const float* __restrict__ in,
                                    __bf16* __restrict__ out, int K, int N) {
  int k = blockIdx.x * blockDim.x + threadIdx.x;
  int n = blockIdx.y;
  if (k < K) out[(size_t)n * K + k] = f2bf(in[(size_t)k * N + n]);
}

// Prep 2: gathered embeddings in bf16, laid out [t][b][e] for direct
// A-fragment loads in layer 0. Fully parallel; keeps all f32->bf16 VALU
// conversion out of the 256-step sequential critical path.
__global__ __launch_bounds__(256) void gather_embed_bf16(
    const int* __restrict__ tokens, const float* __restrict__ emb,
    __bf16* __restrict__ xbf) {
  const int e = threadIdx.x;      // 0..255
  const int b = blockIdx.x;       // 0..255
  const int t = blockIdx.y;       // 0..255
  const int tok = tokens[b * T_ + t];
  xbf[((size_t)t * B_ + b) * E_ + e] = f2bf(emb[(size_t)tok * E_ + e]);
}

__global__ void init_state(float* c0, float* c1, __bf16* h0, __bf16* h1) {
  int i = blockIdx.x * blockDim.x + threadIdx.x;
  if (i < B_ * U_) {
    c0[i] = 0.0f; c1[i] = 0.0f;
    h0[i] = f2bf(0.0f); h1[i] = f2bf(0.0f);
  }
}

// --------------------------------------------------------------------------
// Layer-0 step: z = x_t @ W0^T + h0 @ U0^T ; gates ; write h0', c0.
// Grid: 32 blocks (2 M-blocks x 16 U-chunks), 256 threads = 8 waves.
// Wave owns 32 rows x 16 u-cols and all 4 gate columns (u, u+512, u+1024,
// u+1536) so i/f/g/o land in the same lane/slot -> in-register cell update.
__global__ __launch_bounds__(256) void lstm_l0_step(
    const __bf16* __restrict__ xbf,
    const __bf16* __restrict__ W0t, const __bf16* __restrict__ U0t,
    const float* __restrict__ b0,
    const __bf16* __restrict__ h0r, __bf16* __restrict__ h0w,
    float* __restrict__ c0, int t) {
  const int lane = threadIdx.x & 31;
  const int w    = threadIdx.x >> 5;
  const int wm   = w & 3;            // 4 waves over M
  const int wu   = w >> 2;           // 2 waves over u
  const int mB   = blockIdx.x & 1;
  const int uB   = blockIdx.x >> 1;
  const int r0   = mB * 128 + wm * 32;
  const int u0   = uB * 32 + wu * 16;
  const int ln   = lane & 15;
  const int kh   = lane >> 4;

  v8f acc[4][2];
#pragma unroll
  for (int g = 0; g < 4; ++g)
#pragma unroll
    for (int m = 0; m < 2; ++m) acc[g][m] = v8f{};

  const int row0 = r0 + ln;
  const int row1 = r0 + 16 + ln;

  // Pass 1: x_t @ W0^T (K = 256), pure bf16 fragment loads.
  const __bf16* x0 = xbf + ((size_t)t * B_ + row0) * E_;
  const __bf16* x1 = xbf + ((size_t)t * B_ + row1) * E_;
  for (int k0 = 0; k0 < E_; k0 += 32) {
    const int ka = k0 + kh * 16;
    v16bf a0 = load_a_bf(x0 + ka);
    v16bf a1 = load_a_bf(x1 + ka);
    const int kb = k0 + kh * 16;
#pragma unroll
    for (int g = 0; g < 4; ++g) {
      const int n = g * U_ + u0 + ln;
      v16bf bm = *(const v16bf*)(W0t + (size_t)n * E_ + kb);
      acc[g][0] = wmma_bf16(a0, bm, acc[g][0]);
      acc[g][1] = wmma_bf16(a1, bm, acc[g][1]);
    }
  }
  // Pass 2: h0 @ U0^T (K = 512), weights L2-resident.
  for (int k0 = 0; k0 < U_; k0 += 32) {
    const int ka = k0 + kh * 16;
    v16bf a0 = load_a_bf(h0r + (size_t)row0 * U_ + ka);
    v16bf a1 = load_a_bf(h0r + (size_t)row1 * U_ + ka);
    const int kb = k0 + kh * 16;
#pragma unroll
    for (int g = 0; g < 4; ++g) {
      const int n = g * U_ + u0 + ln;
      const __bf16* bp = U0t + (size_t)n * U_ + kb;
      v16bf bm = *(const v16bf*)(bp);
      if (k0 + 32 < U_) __builtin_prefetch(bp + 32, 0, 1);  // global_prefetch
      acc[g][0] = wmma_bf16(a0, bm, acc[g][0]);
      acc[g][1] = wmma_bf16(a1, bm, acc[g][1]);
    }
  }

  // Cell update, all gates in-register (C layout: lane -> col ln, rows kh*8+j).
  const int cu = u0 + ln;
  const float bi = b0[cu], bf_ = b0[U_ + cu], bg = b0[2 * U_ + cu], bo = b0[3 * U_ + cu];
#pragma unroll
  for (int mt = 0; mt < 2; ++mt) {
    const int rb = r0 + mt * 16 + kh * 8;
#pragma unroll
    for (int j = 0; j < 8; ++j) {
      const size_t idx = (size_t)(rb + j) * U_ + cu;
      const float zi = acc[0][mt][j] + bi;
      const float zf = acc[1][mt][j] + bf_;
      const float zg = acc[2][mt][j] + bg;
      const float zo = acc[3][mt][j] + bo;
      const float co = c0[idx];
      const float cn = sigf(zf) * co + sigf(zi) * tanhf(zg);
      const float hn = sigf(zo) * tanhf(cn);
      c0[idx] = cn;
      h0w[idx] = f2bf(hn);
    }
  }
}

// --------------------------------------------------------------------------
// Layer-1 step: z = h0_new @ W1^T + h1 @ U1^T + b1 ; gates ; h1', c1, h1_f32.
__global__ __launch_bounds__(256) void lstm_l1_step(
    const __bf16* __restrict__ h0n, const __bf16* __restrict__ W1t,
    const __bf16* __restrict__ h1r, const __bf16* __restrict__ U1t,
    const float* __restrict__ b1,
    float* __restrict__ c1, __bf16* __restrict__ h1w, float* __restrict__ h1f) {
  const int lane = threadIdx.x & 31;
  const int w    = threadIdx.x >> 5;
  const int wm   = w & 3;
  const int wu   = w >> 2;
  const int mB   = blockIdx.x & 1;
  const int uB   = blockIdx.x >> 1;
  const int r0   = mB * 128 + wm * 32;
  const int u0   = uB * 32 + wu * 16;
  const int ln   = lane & 15;
  const int kh   = lane >> 4;

  v8f acc[4][2];
#pragma unroll
  for (int g = 0; g < 4; ++g)
#pragma unroll
    for (int m = 0; m < 2; ++m) acc[g][m] = v8f{};

  const int row0 = r0 + ln;
  const int row1 = r0 + 16 + ln;

  for (int k0 = 0; k0 < U_; k0 += 32) {            // h0_new @ W1^T
    const int ka = k0 + kh * 16;
    v16bf a0 = load_a_bf(h0n + (size_t)row0 * U_ + ka);
    v16bf a1 = load_a_bf(h0n + (size_t)row1 * U_ + ka);
    const int kb = k0 + kh * 16;
#pragma unroll
    for (int g = 0; g < 4; ++g) {
      const int n = g * U_ + u0 + ln;
      const __bf16* bp = W1t + (size_t)n * U_ + kb;
      v16bf bm = *(const v16bf*)(bp);
      if (k0 + 32 < U_) __builtin_prefetch(bp + 32, 0, 1);
      acc[g][0] = wmma_bf16(a0, bm, acc[g][0]);
      acc[g][1] = wmma_bf16(a1, bm, acc[g][1]);
    }
  }
  for (int k0 = 0; k0 < U_; k0 += 32) {            // h1 @ U1^T
    const int ka = k0 + kh * 16;
    v16bf a0 = load_a_bf(h1r + (size_t)row0 * U_ + ka);
    v16bf a1 = load_a_bf(h1r + (size_t)row1 * U_ + ka);
    const int kb = k0 + kh * 16;
#pragma unroll
    for (int g = 0; g < 4; ++g) {
      const int n = g * U_ + u0 + ln;
      const __bf16* bp = U1t + (size_t)n * U_ + kb;
      v16bf bm = *(const v16bf*)(bp);
      if (k0 + 32 < U_) __builtin_prefetch(bp + 32, 0, 1);
      acc[g][0] = wmma_bf16(a0, bm, acc[g][0]);
      acc[g][1] = wmma_bf16(a1, bm, acc[g][1]);
    }
  }

  const int cu = u0 + ln;
  const float bi = b1[cu], bf_ = b1[U_ + cu], bg = b1[2 * U_ + cu], bo = b1[3 * U_ + cu];
#pragma unroll
  for (int mt = 0; mt < 2; ++mt) {
    const int rb = r0 + mt * 16 + kh * 8;
#pragma unroll
    for (int j = 0; j < 8; ++j) {
      const size_t idx = (size_t)(rb + j) * U_ + cu;
      const float zi = acc[0][mt][j] + bi;
      const float zf = acc[1][mt][j] + bf_;
      const float zg = acc[2][mt][j] + bg;
      const float zo = acc[3][mt][j] + bo;
      const float co = c1[idx];
      const float cn = sigf(zf) * co + sigf(zi) * tanhf(zg);
      const float hn = sigf(zo) * tanhf(cn);
      c1[idx] = cn;
      h1w[idx] = f2bf(hn);
      h1f[idx] = hn;
    }
  }
}

// --------------------------------------------------------------------------
__global__ __launch_bounds__(256) void final_dense(const float* __restrict__ h1,
                                                   const float* __restrict__ Wd,
                                                   const float* __restrict__ bd,
                                                   float* __restrict__ out) {
  const int b = threadIdx.x;  // 256 threads, one per batch row
  float s = 0.0f;
  for (int u = 0; u < U_; ++u) s += h1[(size_t)b * U_ + u] * Wd[u];
  s += bd[0];
  out[b] = sigf(s);
}

// --------------------------------------------------------------------------
extern "C" void kernel_launch(void* const* d_in, const int* in_sizes, int n_in,
                              void* d_out, int out_size, void* d_ws, size_t ws_size,
                              hipStream_t stream) {
  (void)in_sizes; (void)n_in; (void)out_size; (void)ws_size;
  const int*   tokens = (const int*)d_in[0];
  const float* emb    = (const float*)d_in[1];
  const float* W0     = (const float*)d_in[2];
  const float* U0     = (const float*)d_in[3];
  const float* b0     = (const float*)d_in[4];
  const float* W1     = (const float*)d_in[5];
  const float* U1     = (const float*)d_in[6];
  const float* b1     = (const float*)d_in[7];
  const float* Wd     = (const float*)d_in[8];
  const float* bd     = (const float*)d_in[9];
  float* out = (float*)d_out;

  // Workspace carve (~41 MB total; 256B aligned chunks).
  char* ws = (char*)d_ws;
  size_t off = 0;
  auto carve = [&](size_t bytes) -> char* {
    char* p = ws + off;
    off = (off + bytes + 255) & ~(size_t)255;
    return p;
  };
  __bf16* U0t  = (__bf16*)carve((size_t)G4 * U_ * 2);
  __bf16* W1t  = (__bf16*)carve((size_t)G4 * U_ * 2);
  __bf16* U1t  = (__bf16*)carve((size_t)G4 * U_ * 2);
  __bf16* W0t  = (__bf16*)carve((size_t)G4 * E_ * 2);
  __bf16* xbf  = (__bf16*)carve((size_t)T_ * B_ * E_ * 2);  // 32 MB gathered emb
  __bf16* h0bf = (__bf16*)carve((size_t)2 * B_ * U_ * 2);   // double buffer
  __bf16* h1bf = (__bf16*)carve((size_t)2 * B_ * U_ * 2);   // double buffer
  float*  c0   = (float*)carve((size_t)B_ * U_ * 4);
  float*  c1   = (float*)carve((size_t)B_ * U_ * 4);
  float*  h1f  = (float*)carve((size_t)B_ * U_ * 4);

  dim3 tb(256);
  // Parallel prep (whole-chip, memory-bound): weight transpose + embedding
  // gather, both converting to bf16 once so the recurrence is conversion-free.
  conv_transpose_bf16<<<dim3(2, G4), tb, 0, stream>>>(U0, U0t, U_, G4);
  conv_transpose_bf16<<<dim3(2, G4), tb, 0, stream>>>(W1, W1t, U_, G4);
  conv_transpose_bf16<<<dim3(2, G4), tb, 0, stream>>>(U1, U1t, U_, G4);
  conv_transpose_bf16<<<dim3(1, G4), tb, 0, stream>>>(W0, W0t, E_, G4);
  gather_embed_bf16<<<dim3(B_, T_), tb, 0, stream>>>(tokens, emb, xbf);
  init_state<<<dim3((B_ * U_ + 255) / 256), tb, 0, stream>>>(c0, c1, h0bf, h1bf);

  const size_t HS = (size_t)B_ * U_;
  for (int t = 0; t < T_; ++t) {
    const int rd = t & 1, wr = rd ^ 1;
    lstm_l0_step<<<dim3(32), tb, 0, stream>>>(xbf, W0t, U0t, b0,
                                              h0bf + rd * HS, h0bf + wr * HS, c0, t);
    lstm_l1_step<<<dim3(32), tb, 0, stream>>>(h0bf + wr * HS, W1t,
                                              h1bf + rd * HS, U1t, b1,
                                              c1, h1bf + wr * HS, h1f);
  }
  final_dense<<<dim3(1), tb, 0, stream>>>(h1f, Wd, bd, out);
}
